// TriangleAttention_18511309046152
// MI455X (gfx1250) — matrix operand order repaired and checked by reading, not compiled, not run on hardware
//
#include <hip/hip_runtime.h>

#define N_RES 384
#define CZ 128
#define HH 4
#define CH 32
#define MPOS (N_RES * N_RES)   // 147456
#define INF_F 1.0e9f
#define LN_EPS 1e-5f
#define WLD 136                // padded LDS stride for transposed 128x128 weight
#define VLD 72                 // padded LDS stride for transposed 32x64 V tile
#define NW  (CZ * CZ)          // 16384

typedef __attribute__((ext_vector_type(16))) __bf16 v16bf;
typedef __attribute__((ext_vector_type(8)))  float  v8f;

union AFrag { v16bf v; uint4 q[2]; };

__device__ inline unsigned short f2bf(float f) {
  unsigned int u = __float_as_uint(f);
  unsigned int r = u + 0x7fffu + ((u >> 16) & 1u);
  return (unsigned short)(r >> 16);
}
__device__ inline float bf2f(unsigned short h) {
  return __uint_as_float(((unsigned int)h) << 16);
}
__device__ inline float fast_rcp(float x) {
#if __has_builtin(__builtin_amdgcn_rcpf)
  return __builtin_amdgcn_rcpf(x);
#else
  return 1.f / x;
#endif
}
__device__ inline unsigned int pkmul_bf16(unsigned int a, unsigned int b) {
  const float lo = bf2f((unsigned short)(a & 0xffffu)) * bf2f((unsigned short)(b & 0xffffu));
  const float hi = bf2f((unsigned short)(a >> 16))     * bf2f((unsigned short)(b >> 16));
  return (unsigned int)f2bf(lo) | ((unsigned int)f2bf(hi) << 16);
}
__device__ inline v8f vzero() {
  v8f v;
  #pragma unroll
  for (int i = 0; i < 8; ++i) v[i] = 0.f;
  return v;
}

// A fragment 16x32 bf16 from row-major [row][k] storage (stride ld ushorts).
__device__ inline v16bf load_a_rm(const unsigned short* s, int ld, int row0, int k0) {
  const int lane = threadIdx.x & 31;
  const int m = lane & 15, hi = (lane >> 4) & 1;
  const unsigned short* p = s + (row0 + m) * ld + k0 + hi * 8;
  AFrag f;
  f.q[0] = *(const uint4*)(p);
  f.q[1] = *(const uint4*)(p + 16);
  return f.v;
}

// B fragment 32x16 bf16 where the source is stored transposed: [n][k] (stride ld).
__device__ inline v16bf load_b_nk(const unsigned short* s, int ld, int n0, int k0) {
  const int lane = threadIdx.x & 31;
  const int n = lane & 15, hi = (lane >> 4) & 1;
  const unsigned short* p = s + (n0 + n) * ld + k0 + hi * 8;
  AFrag f;
  f.q[0] = *(const uint4*)(p);
  f.q[1] = *(const uint4*)(p + 16);
  return f.v;
}

// ---------------------------------------------------------------------------
// Kernel 0: one-shot weight prep — convert fp32 weights to bf16, TRANSPOSED
// [e][c], so per-block staging becomes a pure b128 copy. Also builds the
// zero-padded 16x128 w_bias B tile.
// wt layout: [0..5*NW)   : w_q^T, w_k^T, w_v^T, w_g^T, w_o^T (bf16, [e][c])
//            [5*NW..+2048): w_bias^T padded to 16 rows (bf16, [h][c])
// ---------------------------------------------------------------------------
__global__ __launch_bounds__(256) void k0_prep(
    const float* __restrict__ w_q, const float* __restrict__ w_k,
    const float* __restrict__ w_v, const float* __restrict__ w_g,
    const float* __restrict__ w_o, const float* __restrict__ w_bias,
    unsigned short* __restrict__ wt) {
  const int idx = blockIdx.x * 256 + threadIdx.x;
  if (idx < 5 * NW) {
    const int mi = idx / NW;
    const int r  = idx - mi * NW;
    const int e = r >> 7, c = r & 127;
    const float* src = (mi == 0) ? w_q : (mi == 1) ? w_k : (mi == 2) ? w_v
                      : (mi == 3) ? w_g : w_o;
    wt[idx] = f2bf(src[c * CZ + e]);
  } else if (idx < 5 * NW + 16 * CZ) {
    const int r = idx - 5 * NW;
    const int hrow = r >> 7, c = r & 127;
    wt[idx] = (hrow < HH) ? f2bf(w_bias[c * HH + hrow]) : (unsigned short)0;
  }
}

// ---------------------------------------------------------------------------
// Kernel 1: fused LayerNorm + bias/q/k/v/gating projections (bf16 WMMA GEMMs)
// ---------------------------------------------------------------------------
__global__ __launch_bounds__(256) void k1_ln_proj(
    const float* __restrict__ z, const float* __restrict__ gamma,
    const float* __restrict__ beta, const unsigned short* __restrict__ wt,
    const float* __restrict__ b_g,
    unsigned short* __restrict__ qo, unsigned short* __restrict__ ko,
    unsigned short* __restrict__ vo, unsigned short* __restrict__ go,
    float* __restrict__ tri) {
  __shared__ __align__(16) unsigned short zn[64][CZ];   // bf16 LN result (A source)
  __shared__ __align__(16) unsigned short wl[CZ][WLD];  // transposed weight / output staging
  __shared__ __align__(16) unsigned short wb[16][CZ];   // padded w_bias B tile
  __shared__ float redS[64][4];
  __shared__ float redQ[64][4];
  __shared__ float muS[64], rsS[64];

  const int t     = threadIdx.x;
  const int p0    = blockIdx.x * 64;
  const int i_blk = p0 / N_RES;
  const int j0    = p0 - i_blk * N_RES;
  const int row   = t >> 2;  // 0..63
  const int q4    = t & 3;   // 0..3 (32 channels each)

  // ---- LayerNorm over c_z ----
  float vals[32];
  {
    const float* zr = z + (size_t)(p0 + row) * CZ + q4 * 32;
    float s = 0.f, ss = 0.f;
    #pragma unroll
    for (int c = 0; c < 32; ++c) { float x = zr[c]; vals[c] = x; s += x; ss += x * x; }
    redS[row][q4] = s;
    redQ[row][q4] = ss;
  }
  // copy prepped w_bias tile (pure b128 copy, one per thread)
  for (int idx = t; idx < 256; idx += 256)
    ((uint4*)&wb[0][0])[idx] = ((const uint4*)(wt + 5 * NW))[idx];
  __syncthreads();
  if (q4 == 0) {
    float s  = redS[row][0] + redS[row][1] + redS[row][2] + redS[row][3];
    float ss = redQ[row][0] + redQ[row][1] + redQ[row][2] + redQ[row][3];
    float mu  = s * (1.f / 128.f);
    float var = ss * (1.f / 128.f) - mu * mu;
    muS[row] = mu;
    rsS[row] = rsqrtf(var + LN_EPS);
  }
  __syncthreads();
  {
    const float mu = muS[row], rs = rsS[row];
    unsigned int* znrow = (unsigned int*)&zn[row][q4 * 32];
    #pragma unroll
    for (int c2 = 0; c2 < 16; ++c2) {
      const int cc = q4 * 32 + c2 * 2;
      const float v0 = (vals[c2 * 2]     - mu) * rs * gamma[cc]     + beta[cc];
      const float v1 = (vals[c2 * 2 + 1] - mu) * rs * gamma[cc + 1] + beta[cc + 1];
      znrow[c2] = (unsigned int)f2bf(v0) | ((unsigned int)f2bf(v1) << 16);
    }
  }
  __syncthreads();

  const int wave = t >> 5, lane = t & 31;
  const int n  = lane & 15;
  const int hi = (lane >> 4) & 1;
  const int rt  = wave >> 1;        // row tile 0..3 (16 rows each)
  const int ct0 = (wave & 1) * 4;   // col tiles ct0..ct0+3

  // ---- triangle bias via WMMA (128x4 GEMM padded to x16): waves 0..3 ----
  if (wave < 4) {
    v8f acc = vzero();
    #pragma unroll
    for (int kc = 0; kc < 4; ++kc) {
      v16bf aa = load_a_rm(&zn[0][0], CZ, wave * 16, kc * 32);
      v16bf bb = load_b_nk(&wb[0][0], CZ, 0, kc * 32);
      acc = __builtin_amdgcn_wmma_f32_16x16x32_bf16(false, aa, false, bb,
                                                    (short)0, acc, false, false);
    }
    if (n < HH) {
      #pragma unroll
      for (int r = 0; r < 8; ++r)
        tri[(size_t)n * MPOS + p0 + wave * 16 + r + hi * 8] = acc[r];
    }
  }

  v16bf afr[4];
  #pragma unroll
  for (int kc = 0; kc < 4; ++kc) afr[kc] = load_a_rm(&zn[0][0], CZ, rt * 16, kc * 32);

  unsigned short* dsts[3] = { qo, ko, vo };

  #pragma unroll
  for (int pj = 0; pj < 4; ++pj) {
    __syncthreads();
    // stage prepped transposed bf16 weight: pure b128 global->LDS copy
    const unsigned short* wsrcT = wt + (size_t)pj * NW;
    for (int idx = t; idx < 2048; idx += 256) {
      const int e = idx >> 4, c8 = idx & 15;
      ((uint4*)&wl[e][0])[c8] = ((const uint4*)wsrcT)[idx];
    }
    __syncthreads();

    v8f accs[4];
    #pragma unroll
    for (int c4 = 0; c4 < 4; ++c4) {
      v8f acc = vzero();
      #pragma unroll
      for (int kc = 0; kc < 4; ++kc) {
        v16bf bf = load_b_nk(&wl[0][0], WLD, (ct0 + c4) * 16, kc * 32);
        acc = __builtin_amdgcn_wmma_f32_16x16x32_bf16(false, afr[kc], false, bf,
                                                      (short)0, acc, false, false);
      }
      accs[c4] = acc;
    }
    __syncthreads();  // everyone done reading wl -> reuse as output staging

    unsigned short (*st)[CZ] = (unsigned short(*)[CZ])&wl[0][0];
    #pragma unroll
    for (int c4 = 0; c4 < 4; ++c4) {
      const int col = (ct0 + c4) * 16 + n;
      const float bgv = (pj == 3) ? b_g[col] : 0.f;
      #pragma unroll
      for (int r = 0; r < 8; ++r) {
        float v = accs[c4][r];
        if (pj == 0) v *= 0.17677669529663687f;  // 1/sqrt(32)
        if (pj == 3) v = fast_rcp(1.f + __expf(-(v + bgv)));
        st[rt * 16 + r + hi * 8][col] = f2bf(v);
      }
    }
    __syncthreads();

    // coalesced b128 stores from staged tile
    if (pj == 3) {
      const uint4* s4 = (const uint4*)&st[0][0];
      uint4* d4 = (uint4*)(go + (size_t)p0 * CZ);
      for (int idx = t; idx < 1024; idx += 256) d4[idx] = s4[idx];
    } else {
      unsigned short* dst = dsts[pj];
      const uint4* s4 = (const uint4*)&st[0][0];
      for (int idx = t; idx < 1024; idx += 256) {
        const int rr   = idx >> 4;
        const int col0 = (idx & 15) * 8;
        const int hh = col0 >> 5, d0 = col0 & 31;
        *(uint4*)(dst + (size_t)((i_blk * HH + hh) * N_RES + j0 + rr) * CH + d0) = s4[idx];
      }
    }
  }
}

// ---------------------------------------------------------------------------
// Kernel 2: flash-style attention per (row i, head h, 64-query tile)
// ---------------------------------------------------------------------------
__global__ __launch_bounds__(128) void k2_attn(
    const unsigned short* __restrict__ qw, const unsigned short* __restrict__ kw,
    const unsigned short* __restrict__ vw, const unsigned short* __restrict__ gw,
    const float* __restrict__ tri, const float* __restrict__ mask,
    unsigned short* __restrict__ og) {
  __shared__ __align__(16) unsigned short Qs[64][CH];     // [query][d]
  __shared__ __align__(16) unsigned short Ks[64][CH];     // [key][d]
  __shared__ __align__(16) unsigned short Vt[CH][VLD];    // [d][key]
  __shared__ __align__(16) unsigned short Ps[4][16][64];  // per-wave P tile / O staging
  __shared__ __align__(16) float triS[64][64];            // tri + mask bias tile

  const int t    = threadIdx.x;
  const int wave = t >> 5, lane = t & 31;
  const int n  = lane & 15;
  const int hi = (lane >> 4) & 1;
  const int jt = blockIdx.x;
  const int i  = blockIdx.y;
  const int h  = blockIdx.z;

  const size_t headbase = (size_t)((i * HH + h) * N_RES) * CH;

  {  // Q tile
    const uint4* src = (const uint4*)(qw + headbase + (size_t)(jt * 64) * CH);
    uint4* dst = (uint4*)&Qs[0][0];
    for (int idx = t; idx < 256; idx += 128) dst[idx] = src[idx];
  }
  __syncthreads();
  const v16bf qa = load_a_rm(&Qs[0][0], CH, wave * 16, 0);

  v8f oacc[2] = { vzero(), vzero() };
  float mrun[8], lrun[8];
  #pragma unroll
  for (int r = 0; r < 8; ++r) { mrun[r] = -1e30f; lrun[r] = 0.f; }

  const float* trow_base = tri + (size_t)h * MPOS + (size_t)(jt * 64) * N_RES;
  const float* mrow = mask + i * N_RES;

  for (int kt = 0; kt < 6; ++kt) {
    __syncthreads();
    {  // K tile
      const uint4* src = (const uint4*)(kw + headbase + (size_t)(kt * 64) * CH);
      uint4* dst = (uint4*)&Ks[0][0];
      for (int idx = t; idx < 256; idx += 128) dst[idx] = src[idx];
    }
    {  // V tile transposed
      const uint4* src = (const uint4*)(vw + headbase + (size_t)(kt * 64) * CH);
      for (int idx = t; idx < 256; idx += 128) {
        uint4 vv = src[idx];
        const int key = idx >> 2;
        const int d0  = (idx & 3) * 8;
        const unsigned short* u = (const unsigned short*)&vv;
        #pragma unroll
        for (int j = 0; j < 8; ++j) Vt[d0 + j][key] = u[j];
      }
    }
    {  // tri + mask bias tile
      for (int idx = t; idx < 64 * 16; idx += 128) {
        const int rr = idx >> 4, c4 = idx & 15;
        float4 tv = ((const float4*)(trow_base + (size_t)rr * N_RES + kt * 64))[c4];
        const float4 mv = ((const float4*)(mrow + kt * 64))[c4];
        tv.x += INF_F * (mv.x - 1.f);
        tv.y += INF_F * (mv.y - 1.f);
        tv.z += INF_F * (mv.z - 1.f);
        tv.w += INF_F * (mv.w - 1.f);
        ((float4*)&triS[rr][0])[c4] = tv;
      }
    }
    if (kt < 5) {
      __builtin_prefetch(kw + headbase + (size_t)((kt + 1) * 64) * CH, 0, 1);
      __builtin_prefetch(vw + headbase + (size_t)((kt + 1) * 64) * CH, 0, 1);
    }
    __syncthreads();

    // S = Q K^T + bias (bias enters as the WMMA C matrix)
    v8f s[4];
    #pragma unroll
    for (int nt = 0; nt < 4; ++nt) {
      v8f cinit;
      #pragma unroll
      for (int r = 0; r < 8; ++r) cinit[r] = triS[wave * 16 + r + hi * 8][nt * 16 + n];
      v16bf bf = load_b_nk(&Ks[0][0], CH, nt * 16, 0);
      s[nt] = __builtin_amdgcn_wmma_f32_16x16x32_bf16(false, qa, false, bf,
                                                      (short)0, cinit, false, false);
    }

    // online softmax in registers
    float alpha[8];
    #pragma unroll
    for (int r = 0; r < 8; ++r) {
      float mt = fmaxf(fmaxf(s[0][r], s[1][r]), fmaxf(s[2][r], s[3][r]));
      mt = fmaxf(mt, __shfl_xor(mt, 1, 32));
      mt = fmaxf(mt, __shfl_xor(mt, 2, 32));
      mt = fmaxf(mt, __shfl_xor(mt, 4, 32));
      mt = fmaxf(mt, __shfl_xor(mt, 8, 32));
      const float mnew = fmaxf(mrun[r], mt);
      alpha[r] = __expf(mrun[r] - mnew);
      const float e0 = __expf(s[0][r] - mnew);
      const float e1 = __expf(s[1][r] - mnew);
      const float e2 = __expf(s[2][r] - mnew);
      const float e3 = __expf(s[3][r] - mnew);
      s[0][r] = e0; s[1][r] = e1; s[2][r] = e2; s[3][r] = e3;
      float sum = (e0 + e1) + (e2 + e3);
      sum += __shfl_xor(sum, 1, 32);
      sum += __shfl_xor(sum, 2, 32);
      sum += __shfl_xor(sum, 4, 32);
      sum += __shfl_xor(sum, 8, 32);
      lrun[r] = alpha[r] * lrun[r] + sum;
      mrun[r] = mnew;
    }

    // P tile to this wave's private LDS slice
    #pragma unroll
    for (int nt = 0; nt < 4; ++nt) {
      #pragma unroll
      for (int r = 0; r < 8; ++r)
        Ps[wave][r + hi * 8][nt * 16 + n] = f2bf(s[nt][r]);
    }
    __builtin_amdgcn_wave_barrier();

    #pragma unroll
    for (int r = 0; r < 8; ++r) { oacc[0][r] *= alpha[r]; oacc[1][r] *= alpha[r]; }
    #pragma unroll
    for (int kc = 0; kc < 2; ++kc) {
      v16bf pa = load_a_rm(&Ps[wave][0][0], 64, 0, kc * 32);
      #pragma unroll
      for (int dt = 0; dt < 2; ++dt) {
        v16bf vb = load_b_nk(&Vt[0][0], VLD, dt * 16, kc * 32);
        oacc[dt] = __builtin_amdgcn_wmma_f32_16x16x32_bf16(false, pa, false, vb,
                                                           (short)0, oacc[dt], false, false);
      }
    }
  }

  // epilogue: normalize into LDS (reusing P slice), then gated coalesced stores
  const int jq0 = jt * 64 + wave * 16;
  unsigned short (*Ow)[32] = (unsigned short(*)[32])&Ps[wave][0][0];
  #pragma unroll
  for (int r = 0; r < 8; ++r) {
    const float inv = fast_rcp(lrun[r]);
    Ow[r + hi * 8][n]      = f2bf(oacc[0][r] * inv);
    Ow[r + hi * 8][16 + n] = f2bf(oacc[1][r] * inv);
  }
  __builtin_amdgcn_wave_barrier();
  {
    const int orow = lane >> 1;   // 0..15
    const int half = lane & 1;    // 0..1
    const size_t prow = (size_t)(i * N_RES + jq0 + orow) * CZ + h * CH + half * 16;
    const uint4 g0 = *(const uint4*)(gw + prow);
    const uint4 g1 = *(const uint4*)(gw + prow + 8);
    const uint4 o0 = *(const uint4*)(&Ow[orow][half * 16]);
    const uint4 o1 = *(const uint4*)(&Ow[orow][half * 16 + 8]);
    uint4 r0, r1;
    r0.x = pkmul_bf16(o0.x, g0.x);
    r0.y = pkmul_bf16(o0.y, g0.y);
    r0.z = pkmul_bf16(o0.z, g0.z);
    r0.w = pkmul_bf16(o0.w, g0.w);
    r1.x = pkmul_bf16(o1.x, g1.x);
    r1.y = pkmul_bf16(o1.y, g1.y);
    r1.z = pkmul_bf16(o1.z, g1.z);
    r1.w = pkmul_bf16(o1.w, g1.w);
    *(uint4*)(og + prow)     = r0;
    *(uint4*)(og + prow + 8) = r1;
  }
}

// ---------------------------------------------------------------------------
// Kernel 3: output projection  out = (o*g) @ w_o + b_o   (fp32 result)
// ---------------------------------------------------------------------------
__global__ __launch_bounds__(256) void k3_outproj(
    const unsigned short* __restrict__ og, const unsigned short* __restrict__ woT,
    const float* __restrict__ b_o, float* __restrict__ out) {
  __shared__ __align__(16) unsigned short xt[64][CZ];
  __shared__ __align__(16) unsigned short wl[CZ][WLD];  // weight / fp32 output staging

  const int t  = threadIdx.x;
  const int p0 = blockIdx.x * 64;

  {
    const uint4* src = (const uint4*)(og + (size_t)p0 * CZ);
    uint4* dst = (uint4*)&xt[0][0];
    for (int idx = t; idx < 1024; idx += 256) dst[idx] = src[idx];
  }
  for (int idx = t; idx < 2048; idx += 256) {
    const int e = idx >> 4, c8 = idx & 15;
    ((uint4*)&wl[e][0])[c8] = ((const uint4*)woT)[idx];
  }
  __syncthreads();

  const int wave = t >> 5, lane = t & 31;
  const int rt  = wave >> 1;
  const int ct0 = (wave & 1) * 4;
  const int n  = lane & 15;
  const int hi = (lane >> 4) & 1;

  v16bf afr[4];
  #pragma unroll
  for (int kc = 0; kc < 4; ++kc) afr[kc] = load_a_rm(&xt[0][0], CZ, rt * 16, kc * 32);

  v8f accs[4];
  #pragma unroll
  for (int c4 = 0; c4 < 4; ++c4) {
    v8f acc = vzero();
    #pragma unroll
    for (int kc = 0; kc < 4; ++kc) {
      v16bf bf = load_b_nk(&wl[0][0], WLD, (ct0 + c4) * 16, kc * 32);
      acc = __builtin_amdgcn_wmma_f32_16x16x32_bf16(false, afr[kc], false, bf,
                                                    (short)0, acc, false, false);
    }
    accs[c4] = acc;
  }
  __syncthreads();  // done reading wl -> reuse as fp32 staging

  float (*st)[CZ] = (float(*)[CZ])&wl[0][0];
  #pragma unroll
  for (int c4 = 0; c4 < 4; ++c4) {
    const int col = (ct0 + c4) * 16 + n;
    const float bo = b_o[col];
    #pragma unroll
    for (int r = 0; r < 8; ++r)
      st[rt * 16 + r + hi * 8][col] = accs[c4][r] + bo;
  }
  __syncthreads();

  {
    const float4* s4 = (const float4*)&st[0][0];
    float4* d4 = (float4*)(out + (size_t)p0 * CZ);
    for (int idx = t; idx < 2048; idx += 256) d4[idx] = s4[idx];
  }
}

// ---------------------------------------------------------------------------
extern "C" void kernel_launch(void* const* d_in, const int* in_sizes, int n_in,
                              void* d_out, int out_size, void* d_ws, size_t ws_size,
                              hipStream_t stream) {
  (void)in_sizes; (void)n_in; (void)out_size; (void)ws_size;

  const float* z      = (const float*)d_in[0];
  const float* mask   = (const float*)d_in[1];
  const float* gamma  = (const float*)d_in[2];
  const float* beta   = (const float*)d_in[3];
  const float* w_bias = (const float*)d_in[4];
  const float* w_q    = (const float*)d_in[5];
  const float* w_k    = (const float*)d_in[6];
  const float* w_v    = (const float*)d_in[7];
  const float* w_g    = (const float*)d_in[8];
  const float* b_g    = (const float*)d_in[9];
  const float* w_o    = (const float*)d_in[10];
  const float* b_o    = (const float*)d_in[11];
  float* out = (float*)d_out;

  char* ws = (char*)d_ws;
  const size_t szHalf = (size_t)MPOS * CZ * sizeof(unsigned short);  // 37.75 MB
  unsigned short* qo = (unsigned short*)(ws);
  unsigned short* ko = (unsigned short*)(ws + szHalf);
  unsigned short* vo = (unsigned short*)(ws + 2 * szHalf);
  unsigned short* go = (unsigned short*)(ws + 3 * szHalf);
  float*          tri = (float*)(ws + 4 * szHalf);
  unsigned short* og  = (unsigned short*)(ws + 4 * szHalf + (size_t)HH * MPOS * sizeof(float));
  unsigned short* wt  = (unsigned short*)(ws + 5 * szHalf + (size_t)HH * MPOS * sizeof(float));

  const int prep_elems = 5 * NW + 16 * CZ;
  k0_prep<<<(prep_elems + 255) / 256, 256, 0, stream>>>(w_q, w_k, w_v, w_g, w_o, w_bias, wt);
  k1_ln_proj<<<MPOS / 64, 256, 0, stream>>>(z, gamma, beta, wt, b_g, qo, ko, vo, go, tri);
  k2_attn<<<dim3(N_RES / 64, N_RES, HH), 128, 0, stream>>>(qo, ko, vo, go, tri, mask, og);
  k3_outproj<<<MPOS / 64, 256, 0, stream>>>(og, wt + 4 * NW, b_o, out);
}
